// DepthSegmNet_6201932776134
// MI455X (gfx1250) — matrix operand order repaired and checked by reading, not compiled, not run on hardware
//
#include <hip/hip_runtime.h>

// ============================================================================
// DepthSegmNet for MI455X (gfx1250, wave32, WMMA).
// Convs = implicit GEMM on v_wmma_f32_16x16x32_bf16, fp32 accumulation,
// bias+BN+ReLU+residual and 2x nearest-upsample fused into the conv kernel
// (templated over ksize / up2 / K%32==0). Similarity volume uses WMMA (K=64)
// with a wave-private LDS transpose (s_wait_dscnt, no workgroup barriers) and
// a branchless min/max top-3.
// ============================================================================

typedef __attribute__((ext_vector_type(16))) __bf16 v16bf;
typedef __attribute__((ext_vector_type(8)))  float  v8f;

#define BATCH 8
#define SEGP  576   // 24*24 pixels at the similarity level
#define SEGC  64    // feature channels at the similarity level

// ---------------------------------------------------------------------------
// Pack conv weights fp32 [cout][cin*k*k] -> bf16 [cout_pad16][kpad32] (zero
// padded) and fold conv-bias + BatchNorm(eval, mean=0, var=1):
//   y = scale*conv + shift, scale = g/sqrt(1+eps), shift = scale*b + beta.
// ---------------------------------------------------------------------------
__global__ void pack_weights(const float* __restrict__ w, const float* __restrict__ bias,
                             const float* __restrict__ g, const float* __restrict__ be,
                             __bf16* __restrict__ wp, float* __restrict__ scale,
                             float* __restrict__ shift,
                             int cout, int coutp, int kelem, int kpad)
{
    int i = blockIdx.x * blockDim.x + threadIdx.x;
    int total = coutp * kpad;
    if (i < total) {
        int n = i / kpad, k = i - n * kpad;
        float v = (n < cout && k < kelem) ? w[(size_t)n * kelem + k] : 0.f;
        wp[i] = (__bf16)v;
    }
    if (i < cout) {
        float s = g[i] * rsqrtf(1.f + 1e-5f);
        scale[i] = s;
        shift[i] = s * bias[i] + be[i];
    }
}

// ---------------------------------------------------------------------------
// Nearest-downsample mask_train / test_dist 384x384 -> 24x24 (src = dst*16).
// ---------------------------------------------------------------------------
__global__ void prep_mask_dist(const float* __restrict__ mask, const float* __restrict__ distf,
                               float* __restrict__ mpos, float* __restrict__ dist)
{
    int idx = blockIdx.x * blockDim.x + threadIdx.x;    // b*576 + p
    if (idx >= BATCH * SEGP) return;
    int b = idx / SEGP, p = idx - b * SEGP;
    int y = p / 24, x = p - y * 24;
    size_t s = ((size_t)b * 384 + y * 16) * 384 + x * 16;
    mpos[idx] = mask[s];
    dist[idx] = distf[s];
}

// ---------------------------------------------------------------------------
// L2-normalize 64-ch features along C, repack NCHW fp32 -> [b][p][c] bf16.
// ---------------------------------------------------------------------------
__global__ void normalize_pack(const float* __restrict__ in, __bf16* __restrict__ out)
{
    int idx = blockIdx.x * blockDim.x + threadIdx.x;    // b*576 + p
    if (idx >= BATCH * SEGP) return;
    int b = idx / SEGP, p = idx - b * SEGP;
    const float* src = in + (size_t)b * SEGC * SEGP + p;
    float ss = 0.f;
#pragma unroll 8
    for (int c = 0; c < SEGC; ++c) { float v = src[(size_t)c * SEGP]; ss += v * v; }
    float inv = 1.f / fmaxf(sqrtf(ss), 1e-12f);
    __bf16* dst = out + ((size_t)b * SEGP + p) * SEGC;
#pragma unroll 8
    for (int c = 0; c < SEGC; ++c) dst[c] = (__bf16)(src[(size_t)c * SEGP] * inv);
}

// ---------------------------------------------------------------------------
// Cosine-similarity correlation + masked top-3 mean.
// Wave-private LDS slice (row stride 20 floats: 16B-aligned rows, conflict
// free) -> no workgroup barriers needed; LDS RAW resolved with a wave-local
// s_wait_dscnt 0 (LDS ops are in-order within a wave). Top-3 is a branchless
// min/max sorted insert.
// ---------------------------------------------------------------------------
__global__ void __launch_bounds__(128)
sim_topk_wmma(const __bf16* __restrict__ tn, const __bf16* __restrict__ rn,
              const float* __restrict__ mpos,
              float* __restrict__ pos, float* __restrict__ neg)
{
    __shared__ __align__(16) float smem[4][16][20];
    const int lane = threadIdx.x & 31;
    const int wave = threadIdx.x >> 5;
    const int half = lane >> 4;
    const int l15  = lane & 15;
    const int b    = blockIdx.y;
    const int m0   = (blockIdx.x * 4 + wave) * 16;

    // A fragments for this wave's 16 test pixels (two K-steps of 32)
    v16bf a0, a1;
    const __bf16* arow = tn + ((size_t)b * SEGP + (m0 + l15)) * SEGC;
#pragma unroll
    for (int j = 0; j < 16; ++j) {
        int k = (half << 4) + j;    // contiguous k = 16*half + j (matches B)
        a0[j] = arow[k];
        a1[j] = arow[32 + k];
    }

    float t1 = -1e30f, t2 = -1e30f, t3 = -1e30f;

    for (int nt = 0; nt < SEGP / 16; ++nt) {
        const int n = nt * 16 + l15;
        v16bf b0, b1;
        const __bf16* brow = rn + ((size_t)b * SEGP + n) * SEGC;
#pragma unroll
        for (int j = 0; j < 16; ++j) {                      // contiguous k = 16*half + j (matches A)
            b0[j] = brow[(half << 4) + j];
            b1[j] = brow[32 + (half << 4) + j];
        }
        v8f acc = {};
        acc = __builtin_amdgcn_wmma_f32_16x16x32_bf16(false, a0, false, b0, (short)0, acc, false, false);
        acc = __builtin_amdgcn_wmma_f32_16x16x32_bf16(false, a1, false, b1, (short)0, acc, false, false);
#pragma unroll
        for (int v = 0; v < 8; ++v) smem[wave][(half << 3) + v][l15] = acc[v];
        // wave-local LDS RAW fence: all 32 lanes' ds_stores retired before the
        // cross-lane row reads below (waves never share a slice -> no barrier)
        asm volatile("s_wait_dscnt 0x0" ::: "memory");

        // lane scans row l15 (lanes 0-15: pos mask, lanes 16-31: neg mask)
        const float4* rowv = (const float4*)&smem[wave][l15][0];
        float rv[16];
#pragma unroll
        for (int q = 0; q < 4; ++q) {                       // 4x ds_load_b128
            float4 f = rowv[q];
            rv[4*q+0] = f.x; rv[4*q+1] = f.y; rv[4*q+2] = f.z; rv[4*q+3] = f.w;
        }
#pragma unroll
        for (int c = 0; c < 16; ++c) {
            float mv = mpos[b * SEGP + nt * 16 + c];        // uniform -> s_load
            float w  = half ? (1.f - mv) : mv;
            float s  = rv[c] * w;
            // branchless sorted insert into (t1 >= t2 >= t3)
            float n1 = fmaxf(t1, s);
            float r1 = fminf(t1, s);
            float n2 = fmaxf(t2, r1);
            float r2 = fminf(t2, r1);
            t3 = fmaxf(t3, r2);
            t2 = n2;
            t1 = n1;
        }
        asm volatile("" ::: "memory");  // keep next iter's stores after reads
    }
    float mean = (t1 + t2 + t3) * (1.f / 3.f);
    if (half == 0) pos[b * SEGP + m0 + l15] = mean;
    else           neg[b * SEGP + m0 + l15] = mean;
}

// ---------------------------------------------------------------------------
// segm = [sigmoid(pr-nr), pr, sigmoid(pd-nd), pd, dist]
// (2-way softmax == sigmoid of the logit difference)
// ---------------------------------------------------------------------------
__global__ void assemble_segm(const float* __restrict__ pr, const float* __restrict__ nr,
                              const float* __restrict__ pd, const float* __restrict__ nd,
                              const float* __restrict__ dist, float* __restrict__ segm)
{
    int idx = blockIdx.x * blockDim.x + threadIdx.x;
    if (idx >= BATCH * SEGP) return;
    int b = idx / SEGP, p = idx - b * SEGP;
    float smr = 1.f / (1.f + __expf(nr[idx] - pr[idx]));
    float smd = 1.f / (1.f + __expf(nd[idx] - pd[idx]));
    float* o = segm + (size_t)b * 5 * SEGP;
    o[0 * SEGP + p] = smr;
    o[1 * SEGP + p] = pr[idx];
    o[2 * SEGP + p] = smd;
    o[3 * SEGP + p] = pd[idx];
    o[4 * SEGP + p] = dist[idx];
}

// ---------------------------------------------------------------------------
// Implicit-GEMM conv (SAME pad, stride 1) on bf16 WMMA, templated on:
//   KS    : kernel size (1 or 3) -> 1x1 path has zero per-element branches
//   UP2   : input is the fused 2x nearest upsample of `in` (Hin = H/2)
//   FULLK : cin*KS*KS % 32 == 0 -> no k<kelem guard in the hot loop
// Block = 128 threads = 4 waves = 4 M-tiles sharing one 16-wide cout tile.
// A gathers are lane-coalesced (16 lanes = 16 consecutive pixels of one input
// channel); B = 2x global_load_b128 from pre-packed weights + prefetch of the
// next K-step; epilogue = scale/shift/ReLU/residual fused, 2x b128 stores.
// ---------------------------------------------------------------------------
template <int KS, bool UP2, bool FULLK>
__global__ void __launch_bounds__(128)
conv_wmma(const float* __restrict__ in, const __bf16* __restrict__ wp,
          const float* __restrict__ scale, const float* __restrict__ shift,
          float* __restrict__ out,
          int cin, int cout, int H, int W, int relu, int accum)
{
    const int lane = threadIdx.x & 31;
    const int wave = threadIdx.x >> 5;
    const int half = lane >> 4;
    const int l15  = lane & 15;
    const int b    = blockIdx.z;
    const int HW   = H * W;
    const int m0   = (blockIdx.x * 4 + wave) * 16;
    const int n    = blockIdx.y * 16 + l15;
    const int Hin  = UP2 ? (H >> 1) : H;
    const int Win  = UP2 ? (W >> 1) : W;
    const int kelem = cin * KS * KS;
    const int kpad  = FULLK ? kelem : ((kelem + 31) & ~31);

    const int mpix = m0 + l15;                  // A-row pixel for this lane
    const int oy = mpix / W, ox = mpix - oy * W;
    const float*  __restrict__ inb  = in + (size_t)b * cin * Hin * Win;
    const __bf16* __restrict__ wrow = wp + (size_t)n * kpad;

    v8f acc = {};
    for (int k0 = 0; k0 < kpad; k0 += 32) {
        // ---- A fragment: im2col gather fp32 -> bf16 (contiguous k = 16*half + j, same as B) ----
        v16bf a;
#pragma unroll
        for (int j = 0; j < 16; ++j) {
            const int k = k0 + (half << 4) + j;
            float val;
            if (KS == 1) {
                val = inb[(size_t)k * HW + mpix];           // always in range
            } else {
                val = 0.f;
                if (FULLK || k < kelem) {
                    const int ci = k / 9;
                    const int r  = k - ci * 9;
                    const int ky = r / 3;
                    const int kx = r - ky * 3;
                    const int iy = oy + ky - 1;
                    const int ix = ox + kx - 1;
                    if (iy >= 0 && iy < H && ix >= 0 && ix < W) {
                        const int sy = UP2 ? (iy >> 1) : iy;   // fused nearest-up2
                        const int sx = UP2 ? (ix >> 1) : ix;
                        val = inb[((size_t)ci * Hin + sy) * Win + sx];
                    }
                }
            }
            a[j] = (__bf16)val;
        }
        // ---- B fragment: 2x b128 from packed [cout16][kpad] ----
        v16bf bv;
        const __bf16* __restrict__ wseg = wrow + k0 + (half << 4);
#pragma unroll
        for (int j = 0; j < 16; ++j) bv[j] = wseg[j];
        __builtin_prefetch(wrow + k0 + 32, 0, 1);           // global_prefetch_b8

        acc = __builtin_amdgcn_wmma_f32_16x16x32_bf16(false, a, false, bv, (short)0, acc, false, false);
    }

    // ---- fused epilogue: BN scale/shift, ReLU, residual accumulate ----
    if (n < cout) {
        const float sc = scale[n], sh = shift[n];
        float* __restrict__ ob = out + ((size_t)b * cout + n) * HW + m0 + (half << 3);
        float r[8];
#pragma unroll
        for (int v = 0; v < 8; ++v) {       // rows m = m0 + half*8 + v (ISA C layout)
            float y = acc[v] * sc + sh;
            if (relu) y = fmaxf(y, 0.f);
            r[v] = y;
        }
        float4* obv = (float4*)ob;          // 32B-aligned: m0+half*8 ≡ 0 mod 8
        if (accum) {
            float4 o0 = obv[0], o1 = obv[1];
            r[0] += o0.x; r[1] += o0.y; r[2] += o0.z; r[3] += o0.w;
            r[4] += o1.x; r[5] += o1.y; r[6] += o1.z; r[7] += o1.w;
        }
        obv[0] = make_float4(r[0], r[1], r[2], r[3]);
        obv[1] = make_float4(r[4], r[5], r[6], r[7]);
    }
}

// ===========================================================================
// Host orchestration
// ===========================================================================
extern "C" void kernel_launch(void* const* d_in, const int* in_sizes, int n_in,
                              void* d_out, int out_size, void* d_ws, size_t ws_size,
                              hipStream_t stream)
{
    (void)in_sizes; (void)n_in; (void)out_size; (void)ws_size;

    const float* ftr0 = (const float*)d_in[0];
    const float* ftr1 = (const float*)d_in[1];
    const float* ftr2 = (const float*)d_in[2];
    const float* ftr3 = (const float*)d_in[3];
    const float* ftd0 = (const float*)d_in[4];
    const float* ftd1 = (const float*)d_in[5];
    const float* ftd2 = (const float*)d_in[6];
    const float* ftd3 = (const float*)d_in[7];
    const float* trr3 = (const float*)d_in[8];
    const float* trd3 = (const float*)d_in[9];
    const float* maskt = (const float*)d_in[10];
    const float* tdist = (const float*)d_in[11];

    // layer_defs insertion order (cout, cin, k)
    struct LayerDef { int cout, cin, k; };
    static const LayerDef L[18] = {
        {64,1024,1},  // 0  seg0_rgb
        {64,  64,3},  // 1  seg1_rgb
        {64,  64,1},  // 2  seg0_d
        {64,  64,3},  // 3  seg1_d
        {64,   5,3},  // 4  mixer
        {32,  64,3},  // 5  s3
        {32,  32,3},  // 6  s2
        {16,  16,3},  // 7  s1
        { 4,   4,3},  // 8  s0
        {32, 512,3},  // 9  f2
        {16, 256,3},  // 10 f1
        { 4,  64,3},  // 11 f0
        {32,  32,3},  // 12 d2
        {16,  16,3},  // 13 d1
        { 4,   4,3},  // 14 d0
        {16,  32,3},  // 15 post2
        { 4,  16,3},  // 16 post1
        { 2,   4,3},  // 17 post0
    };

    char* ws = (char*)d_ws;
    size_t off = 0;
    auto alloc = [&](size_t bytes) -> void* {
        off = (off + 255) & ~(size_t)255;
        void* p = ws + off;
        off += bytes;
        return p;
    };

    // ---- pack all weights / fold BN ----
    __bf16* wpk[18]; float* wsc[18]; float* wsh[18];
    for (int li = 0; li < 18; ++li) {
        const int kel  = L[li].cin * L[li].k * L[li].k;
        const int kpad = (kel + 31) & ~31;
        const int cp   = (L[li].cout + 15) & ~15;
        wpk[li] = (__bf16*)alloc((size_t)cp * kpad * sizeof(__bf16));
        wsc[li] = (float*)alloc((size_t)L[li].cout * sizeof(float));
        wsh[li] = (float*)alloc((size_t)L[li].cout * sizeof(float));
        const float* w  = (const float*)d_in[12 + li * 4 + 0];
        const float* bb = (const float*)d_in[12 + li * 4 + 1];
        const float* g  = (const float*)d_in[12 + li * 4 + 2];
        const float* be = (const float*)d_in[12 + li * 4 + 3];
        const int total = cp * kpad;
        pack_weights<<<(total + 255) / 256, 256, 0, stream>>>(
            w, bb, g, be, wpk[li], wsc[li], wsh[li], L[li].cout, cp, kel, kpad);
    }

    // ---- activation / intermediate buffers ----
    float*  tA    = (float*)alloc((size_t)BATCH * 64 * SEGP * 4);
    float*  tB    = (float*)alloc((size_t)BATCH * 64 * SEGP * 4);
    __bf16* tn_r  = (__bf16*)alloc((size_t)BATCH * SEGP * SEGC * 2);
    __bf16* rn_r  = (__bf16*)alloc((size_t)BATCH * SEGP * SEGC * 2);
    __bf16* tn_d  = (__bf16*)alloc((size_t)BATCH * SEGP * SEGC * 2);
    __bf16* rn_d  = (__bf16*)alloc((size_t)BATCH * SEGP * SEGC * 2);
    float*  mposb = (float*)alloc((size_t)BATCH * SEGP * 4);
    float*  distb = (float*)alloc((size_t)BATCH * SEGP * 4);
    float*  pos_r = (float*)alloc((size_t)BATCH * SEGP * 4);
    float*  neg_r = (float*)alloc((size_t)BATCH * SEGP * 4);
    float*  pos_d = (float*)alloc((size_t)BATCH * SEGP * 4);
    float*  neg_d = (float*)alloc((size_t)BATCH * SEGP * 4);
    float*  segm  = (float*)alloc((size_t)BATCH * 5 * SEGP * 4);
    float*  mixo  = (float*)alloc((size_t)BATCH * 64 * SEGP * 4);
    float*  s3o   = (float*)alloc((size_t)BATCH * 32 * 2304 * 4);
    float*  sum2  = (float*)alloc((size_t)BATCH * 32 * 2304 * 4);
    float*  p2o   = (float*)alloc((size_t)BATCH * 16 * 9216 * 4);
    float*  sum1  = (float*)alloc((size_t)BATCH * 16 * 9216 * 4);
    float*  p1o   = (float*)alloc((size_t)BATCH * 4 * 36864 * 4);
    float*  sum0  = (float*)alloc((size_t)BATCH * 4 * 36864 * 4);

    auto conv = [&](const float* in, int li, int H, int W, int up2, int relu, int acc, float* out) {
        const LayerDef& l = L[li];
        const int  kel   = l.cin * l.k * l.k;
        const bool fullk = (kel & 31) == 0;
        dim3 grid(H * W / 64, (l.cout + 15) / 16, BATCH);
#define CARGS in, wpk[li], wsc[li], wsh[li], out, l.cin, l.cout, H, W, relu, acc
        if (l.k == 1)            conv_wmma<1, false, true ><<<grid, 128, 0, stream>>>(CARGS);
        else if (!up2 &&  fullk) conv_wmma<3, false, true ><<<grid, 128, 0, stream>>>(CARGS);
        else if (!up2 && !fullk) conv_wmma<3, false, false><<<grid, 128, 0, stream>>>(CARGS);
        else if ( up2 &&  fullk) conv_wmma<3, true , true ><<<grid, 128, 0, stream>>>(CARGS);
        else                     conv_wmma<3, true , false><<<grid, 128, 0, stream>>>(CARGS);
#undef CARGS
    };

    const int EB = (BATCH * SEGP + 255) / 256;

    // ---- stage 0: mask / dist nearest-downsample ----
    prep_mask_dist<<<EB, 256, 0, stream>>>(maskt, tdist, mposb, distb);

    // ---- stage 1: segmentation feature heads + normalize/pack ----
    conv(ftr3, 0, 24, 24, 0, 1, 0, tA);  conv(tA, 1, 24, 24, 0, 0, 0, tB);
    normalize_pack<<<EB, 256, 0, stream>>>(tB, tn_r);
    conv(trr3, 0, 24, 24, 0, 1, 0, tA);  conv(tA, 1, 24, 24, 0, 0, 0, tB);
    normalize_pack<<<EB, 256, 0, stream>>>(tB, rn_r);
    conv(ftd3, 2, 24, 24, 0, 1, 0, tA);  conv(tA, 3, 24, 24, 0, 0, 0, tB);
    normalize_pack<<<EB, 256, 0, stream>>>(tB, tn_d);
    conv(trd3, 2, 24, 24, 0, 1, 0, tA);  conv(tA, 3, 24, 24, 0, 0, 0, tB);
    normalize_pack<<<EB, 256, 0, stream>>>(tB, rn_d);

    // ---- stage 2: WMMA correlation + masked top-3 ----
    sim_topk_wmma<<<dim3(SEGP / 64, BATCH), 128, 0, stream>>>(tn_r, rn_r, mposb, pos_r, neg_r);
    sim_topk_wmma<<<dim3(SEGP / 64, BATCH), 128, 0, stream>>>(tn_d, rn_d, mposb, pos_d, neg_d);
    assemble_segm<<<EB, 256, 0, stream>>>(pos_r, neg_r, pos_d, neg_d, distb, segm);

    // ---- stage 3: decoder (up2 fused into conv input gathers) ----
    conv(segm, 4,  24,  24, 0, 1, 0, mixo);   // mixer
    conv(mixo, 5,  48,  48, 1, 1, 0, s3o);    // s3 on up2(mixer)
    conv(ftr2, 9,  48,  48, 0, 1, 0, sum2);   // f2
    conv(ftd2, 12, 48,  48, 0, 1, 1, sum2);   // + d2
    conv(s3o,  6,  48,  48, 0, 1, 1, sum2);   // + s2
    conv(sum2, 15, 96,  96, 1, 1, 0, p2o);    // post2 on up2(sum)
    conv(ftr1, 10, 96,  96, 0, 1, 0, sum1);   // f1
    conv(ftd1, 13, 96,  96, 0, 1, 1, sum1);   // + d1
    conv(p2o,  7,  96,  96, 0, 1, 1, sum1);   // + s1
    conv(sum1, 16, 192, 192, 1, 1, 0, p1o);   // post1 on up2(sum)
    conv(ftr0, 11, 192, 192, 0, 1, 0, sum0);  // f0
    conv(ftd0, 14, 192, 192, 0, 1, 1, sum0);  // + d0
    conv(p1o,  8,  192, 192, 0, 1, 1, sum0);  // + s0
    conv(sum0, 17, 384, 384, 1, 0, 0, (float*)d_out);  // post0 (no ReLU) -> [8,2,384,384]
}
